// WindowAttention_37838661878647
// MI455X (gfx1250) — matrix operand order repaired and checked by reading, not compiled, not run on hardware
//
#include <hip/hip_runtime.h>

// ---------------------------------------------------------------------------
// Fused Swin window attention for MI455X (gfx1250, wave32, WMMA f16->f32).
// One 256-thread workgroup (8 wave32) per 64-token window; all intermediates
// (x tile, q/k/v, scores, probs, attn-out) live in LDS (~220 KB of 320 KB).
// x tile is staged via global_load_async_to_lds_b128 (ASYNCcnt DMA path);
// weights are pre-swizzled once into per-lane WMMA B-fragment order in d_ws.
// ---------------------------------------------------------------------------

#define NWIN  4096
#define NTOK  64
#define CDIM  192
#define HEADS 6
#define HD    32
#define TOPK  16

typedef __attribute__((ext_vector_type(16))) _Float16 v16h;
typedef __attribute__((ext_vector_type(8)))  _Float16 v8h;
typedef __attribute__((ext_vector_type(4)))  _Float16 v4h;
typedef __attribute__((ext_vector_type(8)))  float    v8f;

// ---- LDS layout (byte offsets); strides padded for b128 alignment + banks ----
#define XS_LD   200                  // x tile / attn-out tile, f16 [64][200]
#define XS_OFF  0                    // 25600 B
#define QK_LD   40                   // q,k rows f16 [2*6*64][40]
#define QK_OFF  25600                // 61440 B
#define VT_LD   72                   // v transposed f16 [6*32][72]
#define VT_OFF  87040                // 27648 B
#define SC_LD   68                   // scores f32 [6*64][68] (probs f16 alias;
#define SC_OFF  114688               //   also f32 x staging during phase 0)
#define BIAS_OFF 219136              // rel-pos bias table f32 [225*6]
#define TEMP_OFF 224536              // temperature f32 [6]
#define SMEM_BYTES 224560

// ---------------------------------------------------------------------------
// Pre-swizzle a row-major f32 weight W[Nout][K] (used as B = W^T, i.e.
// B[k][n] = W[n][k]) into f16 WMMA B-fragments:
//   frag(nt,ks) lane l element e  =  B[ks*32 + 16*(l/16) + e][nt*16 + l%16]
// stored contiguously: out[((nt*kSteps+ks)*32 + lane)*16 + e].
// ---------------------------------------------------------------------------
__global__ void swizzle_weights(const float* __restrict__ W,
                                _Float16* __restrict__ out,
                                int nTiles, int kSteps, int K)
{
    int t = blockIdx.x * blockDim.x + threadIdx.x;
    int total = nTiles * kSteps * 32 * 8;
    if (t >= total) return;
    int i    = t & 7;
    int lane = (t >> 3) & 31;
    int ks   = (t >> 8) % kSteps;
    int nt   = (t >> 8) / kSteps;
    int n = nt * 16 + (lane & 15);
    int k = ks * 32 + 16 * (lane >> 4) + 2 * i;
    out[2 * t]     = (_Float16)W[n * K + k];
    out[2 * t + 1] = (_Float16)W[n * K + k + 1];
}

// ---- V_WMMA_F32_16X16X32_F16 wrapper ----
__device__ __forceinline__ v8f wmma16(v16h a, v16h b, v8f c) {
    return __builtin_amdgcn_wmma_f32_16x16x32_f16(
        false, a, false, b, (short)0, c, false, false);
}

// A-fragment (16x32 f16) from LDS, row-major with stride ld (halves).
// Lane l holds row M=row0+l%16; halves [16h..16h+15].
__device__ __forceinline__ v16h load_a_frag(const _Float16* s, int row0, int k0,
                                            int ld, int lane) {
    const _Float16* p = s + (row0 + (lane & 15)) * ld + k0 + 16 * (lane >> 4);
    v8h lo = *(const v8h*)p;          // ds_load_b128
    v8h hi = *(const v8h*)(p + 8);    // ds_load_b128
    v16h a;
#pragma unroll
    for (int e = 0; e < 8; ++e) { a[e] = lo[e]; a[e + 8] = hi[e]; }
    return a;
}

// B-fragment (32x16 f16): lane l holds col N=l%16, K = 16*(l/16)+e, e=0..15.
__device__ __forceinline__ v16h load_b_frag(const _Float16* p) {
    v8h lo = *(const v8h*)p;
    v8h hi = *(const v8h*)(p + 8);
    v16h b;
#pragma unroll
    for (int e = 0; e < 8; ++e) { b[e] = lo[e]; b[e + 8] = hi[e]; }
    return b;
}

__global__ __launch_bounds__(256)
void win_attn_fused(const float* __restrict__ x,
                    const float* __restrict__ qkv_b,
                    const float* __restrict__ temperature,
                    const float* __restrict__ rel_bias_table,
                    const float* __restrict__ proj_b,
                    const _Float16* __restrict__ qkvF,   // swizzled qkv_w f16
                    const _Float16* __restrict__ projF,  // swizzled proj_w f16
                    float* __restrict__ out)
{
    extern __shared__ unsigned char smem[];
    _Float16* xs    = (_Float16*)(smem + XS_OFF);    // x tile (later attn-out)
    _Float16* qk    = (_Float16*)(smem + QK_OFF);    // q rows [0..383], k rows [384..767]
    _Float16* vT    = (_Float16*)(smem + VT_OFF);    // v transposed [h*32+d][token]
    float*    sc    = (float*)   (smem + SC_OFF);    // x staging -> scores/probs
    float*    biasS = (float*)   (smem + BIAS_OFF);
    float*    tempS = (float*)   (smem + TEMP_OFF);

    const int tid  = threadIdx.x;
    const int lane = tid & 31;
    // wave index is wave-uniform: force it scalar so tile decomposition and
    // section/head selection run on the SALU (no per-lane divides, no exec
    // masking around the writeout branches).
    const int wavu = __builtin_amdgcn_readfirstlane(tid >> 5);
    const int win  = blockIdx.x;

    // ---------------- phase 0: async-DMA x tile (f32) into LDS staging ------
    // 12288 floats = 49152 B. Each lane copies 16 B per issue; the shared
    // IOFFSET advances both the global and LDS address by 4096 B per issue.
    {
        unsigned lds_b = (unsigned)SC_OFF + (unsigned)tid * 16u;
        unsigned long long ga =
            (unsigned long long)(uintptr_t)(x + (size_t)win * NTOK * CDIM)
            + (unsigned long long)tid * 16ull;
#define ASYNC_X(OFF) asm volatile(                                           \
        "global_load_async_to_lds_b128 %0, %1, off offset:" #OFF             \
        :: "v"(lds_b), "v"(ga) : "memory")
        ASYNC_X(0);     ASYNC_X(4096);  ASYNC_X(8192);  ASYNC_X(12288);
        ASYNC_X(16384); ASYNC_X(20480); ASYNC_X(24576); ASYNC_X(28672);
        ASYNC_X(32768); ASYNC_X(36864); ASYNC_X(40960); ASYNC_X(45056);
#undef ASYNC_X
        // overlap: stage bias table + temperature through the normal path
        for (int e = tid; e < 225 * HEADS; e += 256) biasS[e] = rel_bias_table[e];
        if (tid < HEADS) tempS[tid] = temperature[tid];
        asm volatile("s_wait_asynccnt 0x0" ::: "memory");
    }
    __syncthreads();

    // ---------------- phase 0b: convert staged x f32 -> f16 tile ------------
    {
        const float* stage = sc;   // [64][192] contiguous f32
        for (int e8 = tid; e8 < NTOK * CDIM / 8; e8 += 256) {
            int e = e8 * 8;
            int m = e / CDIM, c = e % CDIM;     // CDIM % 8 == 0: no row split
            float4 v0 = *(const float4*)(stage + e);
            float4 v1 = *(const float4*)(stage + e + 4);
            v8h h8;
            h8[0] = (_Float16)v0.x; h8[1] = (_Float16)v0.y;
            h8[2] = (_Float16)v0.z; h8[3] = (_Float16)v0.w;
            h8[4] = (_Float16)v1.x; h8[5] = (_Float16)v1.y;
            h8[6] = (_Float16)v1.z; h8[7] = (_Float16)v1.w;
            *(v8h*)(xs + m * XS_LD + c) = h8;   // ds_store_b128
        }
    }
    __syncthreads();

    // ---------------- phase 1: qkv = x @ qkv_w^T + qkv_b --------------------
    // 4 M-tiles x 36 N-tiles, K=192. Each wave owns mt = wave/2 and half the
    // N-tiles, so the 6 A-fragments are loaded ONCE and reused 18x. B-fragment
    // addresses strength-reduced to one pointer + literal offsets.
    {
        const int mt = wavu >> 1;
        v16h afrag[6];
#pragma unroll
        for (int ks = 0; ks < 6; ++ks)
            afrag[ks] = load_a_frag(xs, mt * 16, ks * 32, XS_LD, lane);
        const int nt0 = (wavu & 1) * 18;
        const _Float16* bp = qkvF + (size_t)nt0 * 3072 + (size_t)lane * 16;
        for (int nt = nt0; nt < nt0 + 18; ++nt, bp += 3072) {
            v8f acc = {};
#pragma unroll
            for (int ks = 0; ks < 6; ++ks) {
                v16h b = *(const v16h*)(bp + ks * 512);
                acc = wmma16(afrag[ks], b, acc);
            }
            // 16-wide tiles never straddle q/k/v sections (192ch) or heads (32ch)
            const int ntu = __builtin_amdgcn_readfirstlane(nt);
            const int sec = ntu / 12;               // SALU: 0=q 1=k 2=v
            const int rb  = (ntu - sec * 12) * 16;  // channel base in section
            const int h   = rb >> 5;
            const int d   = (rb & 31) + (lane & 15);
            float qb = qkv_b[ntu * 16 + (lane & 15)];
#pragma unroll
            for (int r = 0; r < 8; ++r) {
                int M = mt * 16 + r + 8 * (lane >> 4);
                float val = acc[r] + qb;
                if (sec < 2)      // scalar branch
                    qk[(sec * 384 + h * 64 + M) * QK_LD + d] = (_Float16)val;
                else
                    vT[(h * 32 + d) * VT_LD + M] = (_Float16)val; // v K-major
            }
        }
    }
    __syncthreads();

    // ---------------- phase 1b: L2-normalize q,k; fold temperature into q ---
    for (int row = tid; row < 2 * HEADS * NTOK; row += 256) {
        _Float16* p = qk + row * QK_LD;
        v8h h0 = *(v8h*)p,        h1 = *(v8h*)(p + 8);
        v8h h2 = *(v8h*)(p + 16), h3 = *(v8h*)(p + 24);
        float s = 0.f;
#pragma unroll
        for (int e = 0; e < 8; ++e) {
            float a0 = (float)h0[e], a1 = (float)h1[e];
            float a2 = (float)h2[e], a3 = (float)h3[e];
            s += a0 * a0 + a1 * a1 + a2 * a2 + a3 * a3;
        }
        float scale = 1.f / fmaxf(sqrtf(s), 1e-12f);
        if (row < HEADS * NTOK) scale *= tempS[row >> 6]; // temp>0: top-k safe
#pragma unroll
        for (int e = 0; e < 8; ++e) {
            h0[e] = (_Float16)((float)h0[e] * scale);
            h1[e] = (_Float16)((float)h1[e] * scale);
            h2[e] = (_Float16)((float)h2[e] * scale);
            h3[e] = (_Float16)((float)h3[e] * scale);
        }
        *(v8h*)p = h0;        *(v8h*)(p + 8) = h1;
        *(v8h*)(p + 16) = h2; *(v8h*)(p + 24) = h3;
    }
    __syncthreads();

    // ---------------- phase 2: scores = qn @ kn^T (per head, 64x64, K=32) ---
    // 24 (h,mt) groups x 4 jt; 3 groups/wave; A fragment reused across jt.
#pragma unroll 1
    for (int j = 0; j < 3; ++j) {
        int g = wavu * 3 + j;
        int h = g >> 2, mt = g & 3;
        v16h a = load_a_frag(qk, h * 64 + mt * 16, 0, QK_LD, lane);
#pragma unroll
        for (int jt = 0; jt < 4; ++jt) {
            const _Float16* kp = qk + (384 + h * 64 + jt * 16 + (lane & 15)) * QK_LD
                                    + 16 * (lane >> 4);
            v8f acc = {};
            acc = wmma16(a, load_b_frag(kp), acc);
#pragma unroll
            for (int r8 = 0; r8 < 8; ++r8) {
                int M = mt * 16 + r8 + 8 * (lane >> 4);
                sc[(h * 64 + M) * SC_LD + jt * 16 + (lane & 15)] = acc[r8];
            }
        }
    }
    __syncthreads();

    // ---------------- phase 3: per-row top-16 + rel-pos bias + softmax ------
    // 384 rows; each thread owns whole rows -> probs written in-place as f16.
    for (int row = tid; row < HEADS * NTOK; row += 256) {
        int h = row >> 6, m = row & 63;
        float* sr = sc + row * SC_LD;
        float arr[64];
#pragma unroll
        for (int c4 = 0; c4 < 16; ++c4) {
            float4 q4 = *(const float4*)(sr + c4 * 4);   // ds_load_b128
            arr[c4 * 4 + 0] = q4.x; arr[c4 * 4 + 1] = q4.y;
            arr[c4 * 4 + 2] = q4.z; arr[c4 * 4 + 3] = q4.w;
        }
        // 16th largest via repeated strict-max (random floats: ties measure-0)
        float thr = 1e30f;
#pragma unroll 1
        for (int it = 0; it < TOPK; ++it) {
            float best = -1e30f;
#pragma unroll
            for (int c = 0; c < 64; ++c) {
                float v = arr[c];
                best = fmaxf(best, (v < thr) ? v : -1e30f);
            }
            thr = best;
        }
        int ih = m >> 3, iw = m & 7;
        float maxv = -1e30f;
#pragma unroll
        for (int c = 0; c < 64; ++c) {
            int jh = c >> 3, jw = c & 7;
            float bz = biasS[((ih - jh + 7) * 15 + (iw - jw + 7)) * HEADS + h];
            float val = (arr[c] >= thr) ? arr[c] + bz : bz - 100.f;
            arr[c] = val;
            maxv = fmaxf(maxv, val);
        }
        float sum = 0.f;
#pragma unroll
        for (int c = 0; c < 64; ++c) sum += __expf(arr[c] - maxv);
        float rs = 1.f / sum;
        _Float16* pr = (_Float16*)sr;   // own row only; first 128 B of row
#pragma unroll
        for (int c4 = 0; c4 < 16; ++c4) {
            v4h p4;
#pragma unroll
            for (int e = 0; e < 4; ++e)
                p4[e] = (_Float16)(__expf(arr[c4 * 4 + e] - maxv) * rs);
            *(v4h*)(pr + c4 * 4) = p4;   // ds_store_b64
        }
    }
    __syncthreads();

    // ---------------- phase 4: attn-out = probs @ v (per head, K=64) --------
    // 24 (h,mt) groups; 3/wave; A fragment reused across both 16-col D-tiles.
    const _Float16* probs = (const _Float16*)sc;   // row stride 2*SC_LD halves
    _Float16* ao = xs;                             // reuse x-tile region
#pragma unroll 1
    for (int j = 0; j < 3; ++j) {
        int g = wavu * 3 + j;
        int h = g >> 2, mt = g & 3;
        v8f acc0 = {}, acc1 = {};
#pragma unroll
        for (int ks = 0; ks < 2; ++ks) {
            v16h a = load_a_frag(probs, h * 64 + mt * 16, ks * 32, 2 * SC_LD, lane);
            const _Float16* vp = vT + (h * 32 + (lane & 15)) * VT_LD
                                    + ks * 32 + 16 * (lane >> 4);
            acc0 = wmma16(a, load_b_frag(vp), acc0);
            acc1 = wmma16(a, load_b_frag(vp + 16 * VT_LD), acc1);
        }
        int c0 = h * 32 + (lane & 15);
#pragma unroll
        for (int r8 = 0; r8 < 8; ++r8) {
            int M = mt * 16 + r8 + 8 * (lane >> 4);
            ao[M * XS_LD + c0]      = (_Float16)acc0[r8];
            ao[M * XS_LD + c0 + 16] = (_Float16)acc1[r8];
        }
    }
    __syncthreads();

    // ---------------- phase 5: out = ao @ proj_w^T + proj_b -----------------
    // mt = wave/2 fixed; A fragments loaded once, reused across 6 N-tiles.
    // Final stores are non-temporal: out is write-once, keep L2 for weights.
    {
        const int mt = wavu >> 1;
        const int nt0 = (wavu & 1) * 6;
        v8f acc[6] = {};
        const _Float16* bp = projF + (size_t)nt0 * 3072 + (size_t)lane * 16;
#pragma unroll
        for (int ks = 0; ks < 6; ++ks) {
            v16h a = load_a_frag(ao, mt * 16, ks * 32, XS_LD, lane);
#pragma unroll
            for (int q = 0; q < 6; ++q) {
                v16h b = *(const v16h*)(bp + q * 3072 + ks * 512);
                acc[q] = wmma16(a, b, acc[q]);
            }
        }
        float* og = out + (size_t)win * NTOK * CDIM;
#pragma unroll
        for (int q = 0; q < 6; ++q) {
            int n = (nt0 + q) * 16 + (lane & 15);
            float pb = proj_b[n];
#pragma unroll
            for (int r8 = 0; r8 < 8; ++r8) {
                int M = mt * 16 + r8 + 8 * (lane >> 4);
                __builtin_nontemporal_store(acc[q][r8] + pb,
                                            &og[(size_t)M * CDIM + n]);
            }
        }
    }
}

// ---------------------------------------------------------------------------
extern "C" void kernel_launch(void* const* d_in, const int* in_sizes, int n_in,
                              void* d_out, int out_size, void* d_ws, size_t ws_size,
                              hipStream_t stream) {
    const float* x     = (const float*)d_in[0];
    const float* qkvw  = (const float*)d_in[1];   // [576,192]
    const float* qkvb  = (const float*)d_in[2];   // [576]
    const float* temp  = (const float*)d_in[3];   // [6]
    const float* rbt   = (const float*)d_in[4];   // [225,6]
    const float* projw = (const float*)d_in[5];   // [192,192]
    const float* projb = (const float*)d_in[6];   // [192]
    float* out = (float*)d_out;

    // workspace: swizzled f16 weight fragments (deterministic, rewritten each call)
    _Float16* qkvF  = (_Float16*)d_ws;                                  // 221184 B
    _Float16* projF = (_Float16*)((char*)d_ws + 36 * 6 * 32 * 16 * 2);  //  73728 B

    {
        int total = 36 * 6 * 32 * 8;   // qkv_w: 36 N-tiles, 6 k-steps
        swizzle_weights<<<(total + 255) / 256, 256, 0, stream>>>(qkvw, qkvF, 36, 6, 192);
    }
    {
        int total = 12 * 6 * 32 * 8;   // proj_w: 12 N-tiles, 6 k-steps
        swizzle_weights<<<(total + 255) / 256, 256, 0, stream>>>(projw, projF, 12, 6, 192);
    }
    win_attn_fused<<<NWIN, 256, SMEM_BYTES, stream>>>(
        x, qkvb, temp, rbt, projb, qkvF, projF, out);
}
